// AttentionLayer_4544075399227
// MI455X (gfx1250) — matrix operand order repaired and checked by reading, not compiled, not run on hardware
//
#include <hip/hip_runtime.h>
#include <hip/hip_bf16.h>

typedef __attribute__((ext_vector_type(16))) _Float16 v16h;
typedef __attribute__((ext_vector_type(8)))  float    v8f;

// Problem constants (fixed by the reference)
#define BB 2
#define SS 512
#define DD 256
#define TQ 16          // query rows per block
#define NTHREADS 256   // 8 waves of 32

__global__ __launch_bounds__(NTHREADS)
void manhattan_attn_kernel(const float* __restrict__ hin, float* __restrict__ outp) {
    // LDS carve-out (64 KB total):
    //  smem_q : 16x256 f32 query tile                      (16 KB, live whole kernel)
    //  smem_s : 16x512 f32 scores  -> reused as 32x256 f32 B-stage in GEMM (32 KB)
    //  smem_k : 16x256 f32 key tile -> reused as 16x512 f16 probs         (16 KB)
    __shared__ __align__(16) char smem_q[TQ * DD * 4];
    __shared__ __align__(16) char smem_s[TQ * SS * 4];
    __shared__ __align__(16) char smem_k[TQ * DD * 4];

    float*     qtile  = (float*)smem_q;
    float*     scores = (float*)smem_s;
    float*     bstage = (float*)smem_s;   // alias (after softmax)
    float*     ktile  = (float*)smem_k;
    _Float16*  pf16   = (_Float16*)smem_k; // alias (after distance phase)

    const int tid  = threadIdx.x;
    const int lane = tid & 31;
    const int wid  = tid >> 5;

    const int b  = blockIdx.x / (SS / TQ);
    const int i0 = (blockIdx.x % (SS / TQ)) * TQ;

    const float* hb = hin + (size_t)b * SS * DD;

    // ---------------- Phase 1: load Q tile (coalesced float4) ----------------
    {
        const float4* src = (const float4*)(hb + i0 * DD);
        float4*       dst = (float4*)qtile;
        for (int idx = tid; idx < (TQ * DD) / 4; idx += NTHREADS) dst[idx] = src[idx];
    }
    __syncthreads();

    // ---------------- Phase 2: Manhattan distances -> scores[16][512] -------
    const int ti = tid >> 4;   // query row 0..15
    const int tj = tid & 15;   // key row within tile 0..15
    for (int jt = 0; jt < SS / TQ; ++jt) {
        // stage key tile (coalesced float4)
        {
            const float4* src = (const float4*)(hb + jt * TQ * DD);
            float4*       dst = (float4*)ktile;
            for (int idx = tid; idx < (TQ * DD) / 4; idx += NTHREADS) dst[idx] = src[idx];
        }
        __syncthreads();

        const float4* q4 = (const float4*)(qtile + ti * DD);
        const float4* k4 = (const float4*)(ktile + tj * DD);
        float s = 0.0f;
        #pragma unroll 8
        for (int d = 0; d < DD / 4; ++d) {
            float4 a = q4[d];
            float4 c = k4[d];
            s += fabsf(a.x - c.x) + fabsf(a.y - c.y) + fabsf(a.z - c.z) + fabsf(a.w - c.w);
        }
        scores[ti * SS + jt * TQ + tj] = s;
        __syncthreads();
    }

    // ---------------- Phase 3: softmax over keys, probs -> f16 --------------
    // 8 waves, 2 rows each; lane covers 16 strided columns.
    for (int rr = 0; rr < 2; ++rr) {
        const int r = wid * 2 + rr;
        float m = -3.0e38f;
        float v[16];
        #pragma unroll
        for (int t = 0; t < 16; ++t) {
            v[t] = -scores[r * SS + lane + t * 32];   // logits = -distance
            m = fmaxf(m, v[t]);
        }
        #pragma unroll
        for (int off = 16; off > 0; off >>= 1) m = fmaxf(m, __shfl_xor(m, off, 32));
        float sum = 0.0f;
        #pragma unroll
        for (int t = 0; t < 16; ++t) { v[t] = __expf(v[t] - m); sum += v[t]; }
        #pragma unroll
        for (int off = 16; off > 0; off >>= 1) sum += __shfl_xor(sum, off, 32);
        const float inv = 1.0f / sum;
        #pragma unroll
        for (int t = 0; t < 16; ++t)
            pf16[r * SS + lane + t * 32] = (_Float16)(v[t] * inv);
    }

    // ---------------- Phase 4: context = P (16x512) @ H (512x256) via WMMA --
    // Each wave owns two 16x16 output tiles at columns n0 and n0+16.
    const int n0   = wid * 32;
    const int nn   = lane & 15;         // N within tile / M within tile for A
    const int hi   = lane >> 4;         // lane group 0/1
    v8f acc0 = {};
    v8f acc1 = {};

    for (int kt = 0; kt < SS / 32; ++kt) {
        __syncthreads();
        // stage B chunk: rows kt*32 .. kt*32+31, all 256 cols (coalesced float4)
        {
            const float4* src = (const float4*)(hb + kt * 32 * DD);
            float4*       dst = (float4*)bstage;
            for (int idx = tid; idx < (32 * DD) / 4; idx += NTHREADS) dst[idx] = src[idx];
        }
        __syncthreads();

        // Build A fragment: 16-bit A-matrix 16x32 layout (ISA 7.12.2)
        // lane L: M = L%16 ; VGPR v holds K = base(v) + 8*(L>=16) + {0,1}
        const _Float16* prow = pf16 + nn * SS + kt * 32;
        v16h a;
        #pragma unroll
        for (int v = 0; v < 8; ++v) {
            const int kb = ((v < 4) ? (2 * v) : (16 + 2 * (v - 4))) + hi * 8;
            a[2 * v]     = prow[kb];
            a[2 * v + 1] = prow[kb + 1];
        }

        // Build B fragments: 16-bit B-matrix 32x16 layout
        // lane L: N = L%16 ; element e (VGPR e/2, half e%2) holds K = 8*(L>=16) + e + (e>=8 ? 8 : 0)
        const int koff = hi * 8;
        v16h b0, b1;
        #pragma unroll
        for (int e = 0; e < 16; ++e) {
            b0[e] = (_Float16)bstage[(koff + e + (e >= 8 ? 8 : 0)) * DD + n0 + nn];
            b1[e] = (_Float16)bstage[(koff + e + (e >= 8 ? 8 : 0)) * DD + n0 + 16 + nn];
        }

        // D = A*B + C  (f16 inputs, f32 accumulate)
        acc0 = __builtin_amdgcn_wmma_f32_16x16x32_f16(
            false, a, false, b0, (short)0, acc0, false, false);
        acc1 = __builtin_amdgcn_wmma_f32_16x16x32_f16(
            false, a, false, b1, (short)0, acc1, false, false);
    }

    // ---------------- Phase 5: write out = [h | context] --------------------
    const size_t bOff = (size_t)b * SS * (2 * DD);

    // h copy from LDS q-tile (coalesced float4)
    for (int idx = tid; idx < (TQ * DD) / 4; idx += NTHREADS) {
        const int r  = idx / (DD / 4);
        const int c4 = idx % (DD / 4);
        float4* dst = (float4*)(outp + bOff + (size_t)(i0 + r) * (2 * DD));
        dst[c4] = ((const float4*)qtile)[idx];
    }

    // context from C/D VGPR layout: VGPR v -> row M = v + 8*hi, col N = lane%16
    #pragma unroll
    for (int v = 0; v < 8; ++v) {
        const int m = v + hi * 8;
        float* orow = outp + bOff + (size_t)(i0 + m) * (2 * DD) + DD;
        orow[n0 + nn]      = acc0[v];
        orow[n0 + 16 + nn] = acc1[v];
    }
}

extern "C" void kernel_launch(void* const* d_in, const int* in_sizes, int n_in,
                              void* d_out, int out_size, void* d_ws, size_t ws_size,
                              hipStream_t stream) {
    (void)in_sizes; (void)n_in; (void)d_ws; (void)ws_size; (void)out_size;
    const float* hin = (const float*)d_in[0];
    float* outp = (float*)d_out;
    dim3 grid(BB * (SS / TQ));   // 64 workgroups
    dim3 block(NTHREADS);        // 8 waves (wave32)
    manhattan_attn_kernel<<<grid, block, 0, stream>>>(hin, outp);
}
